// MultiHeadAttention_14628658610733
// MI455X (gfx1250) — matrix operand order repaired and checked by reading, not compiled, not run on hardware
//
#include <hip/hip_runtime.h>

typedef unsigned short u16;
typedef unsigned int   u32;
typedef unsigned long long u64;
typedef __attribute__((ext_vector_type(16))) __bf16 v16bf;
typedef __attribute__((ext_vector_type(2)))  __bf16 v2bf;
typedef __attribute__((ext_vector_type(8)))  float  v8f;
typedef __attribute__((ext_vector_type(4)))  int    i32x4;
typedef __attribute__((ext_vector_type(8)))  int    i32x8;
typedef __attribute__((ext_vector_type(4)))  unsigned int u32x4;

#define Bsz  2
#define Tlen 2048
#define Cdim 1024
#define Hn   16
#define Dh   64

#define MODE_Q 0
#define MODE_K 1
#define MODE_V 2
#define MODE_O 3

// ---- CDNA5 feature gates (device pass only; host pass takes fallbacks) ----
#if defined(__gfx1250__) && __has_builtin(__builtin_amdgcn_global_load_async_to_lds_b128) && \
    __has_builtin(__builtin_amdgcn_s_wait_asynccnt)
#define USE_ASYNC_LDS 1
typedef __attribute__((address_space(1))) i32x4 gi32x4;   // global int4
typedef __attribute__((address_space(3))) i32x4 li32x4;   // LDS int4
#else
#define USE_ASYNC_LDS 0
#endif

#if defined(__gfx1250__) && __has_builtin(__builtin_amdgcn_tensor_load_to_lds) && \
    __has_builtin(__builtin_amdgcn_s_wait_tensorcnt)
#define USE_TDM 1
#else
#define USE_TDM 0
#endif

#if defined(__gfx1250__) && __has_builtin(__builtin_amdgcn_cvt_pk_bf16_f32)
#define USE_CVT_PK 1
#else
#define USE_CVT_PK 0
#endif

union FragBF { v16bf v; u32 u[8]; };

__device__ __forceinline__ u16 f2bf(float f) {
  u32 x = __float_as_uint(f);
  u32 r = x + 0x7FFFu + ((x >> 16) & 1u);
  return (u16)(r >> 16);
}

__device__ __forceinline__ u32 pack2bf(float f0, float f1) {
#if USE_CVT_PK
  union { v2bf v; u32 u; } c;
  c.v = __builtin_amdgcn_cvt_pk_bf16_f32(f0, f1);
  return c.u;
#else
  return (u32)f2bf(f0) | ((u32)f2bf(f1) << 16);
#endif
}

#if USE_TDM
// LDS byte offset of a generic pointer into shared memory
__device__ __forceinline__ u32 lds_off(const void* p) {
  return (u32)(uintptr_t)(__attribute__((address_space(3))) const void*)p;
}
// Issue a TDM 2D tile load: tile_d0 x tile_d1 elements of 2 bytes,
// global row stride = strideElems, into LDS at ldsaddr (row-contiguous).
// D# layout per CDNA5 ISA.
__device__ __forceinline__ void tdm_load_2d_b16(const void* gaddr, u32 ldsaddr,
                                                u32 tensor_d0, u32 tensor_d1,
                                                u32 tile_d0, u32 tile_d1,
                                                u32 strideElems) {
  const u64 ga = (u64)(uintptr_t)gaddr;
  u32x4 g0;
  g0[0] = 1u;                                    // count=1, user descriptor
  g0[1] = ldsaddr;                               // lds_addr  (bits 63:32)
  g0[2] = (u32)ga;                               // global_addr[31:0]
  g0[3] = (u32)(ga >> 32) | (2u << 30);          // global_addr[56:32] | type=2
  i32x8 g1;
  g1[0] = (int)(1u << 16);                       // data_size=1 (2 bytes)
  g1[1] = (int)((tensor_d0 & 0xFFFFu) << 16);    // tensor_dim0[15:0]
  g1[2] = (int)((tensor_d0 >> 16) | ((tensor_d1 & 0xFFFFu) << 16));
  g1[3] = (int)((tensor_d1 >> 16) | ((tile_d0 & 0xFFFFu) << 16));
  g1[4] = (int)(tile_d1 & 0xFFFFu);              // tile_dim1 | tile_dim2=0
  g1[5] = (int)strideElems;                      // tensor_dim0_stride[31:0]
  g1[6] = 0;                                     // stride0 hi | stride1 lo
  g1[7] = 0;
  i32x4 gz4  = {0, 0, 0, 0};                     // groups 2/3: 2D tensor
  i32x8 gz8  = {0, 0, 0, 0, 0, 0, 0, 0};         // extra group (6-arg form)
  __builtin_amdgcn_tensor_load_to_lds(g0, g1, gz4, gz4, gz8, 0);
}
#endif

// ---------------------------------------------------------------------------
// GEMM  out = A(M x 1024) * W(1024 x 1024) + bias, M = B*T = 4096.
// 128x128 block tile, 8 waves, each wave 32x64 (2x4 WMMA 16x16 tiles).
// A staged to LDS as bf16 [m][k] (async DMA when A already bf16); W staged
// transposed [n][k] so B fragments are contiguous per the CDNA5 B layout.
// ---------------------------------------------------------------------------
template <int A_BF16, int MODE>
__global__ __launch_bounds__(256)
void gemm_bias_kernel(const void* __restrict__ Av,
                      const float* __restrict__ W,
                      const float* __restrict__ bias,
                      void* __restrict__ out)
{
  __shared__ __align__(16) u16 As[128 * 32];
  __shared__ __align__(16) u16 Bs[128 * 32];

  const int tid  = threadIdx.x;
  const int lane = tid & 31;
  const int w    = tid >> 5;
  const int ml   = lane & 15;
  const int hi   = (lane >> 4) & 1;
  const int wm   = (w >> 1) * 32;
  const int wn   = (w & 1) * 64;
  const int n0   = blockIdx.x * 128;
  const int m0   = blockIdx.y * 128;

  v8f acc[2][4] = {};

  const int arow = tid >> 1;          // 0..127
  const int acol = (tid & 1) * 16;    // 0,16
  const int brow = tid >> 3;          // 0..31  (k row of W)
  const int bcol = (tid & 7) * 16;    // 0..112 (n offset)

  for (int kk = 0; kk < Cdim; kk += 32) {
    __syncthreads();
    // ---- stage A tile (128 x 32) as bf16, row-major [m][k]
#if USE_ASYNC_LDS
    if (A_BF16) {   // raw bf16 copy -> async DMA to LDS (ASYNCcnt)
      const u16* gsrc = (const u16*)Av + (size_t)(m0 + arow) * Cdim + kk + acol;
      u16* ldst = &As[arow * 32 + acol];
      __builtin_amdgcn_global_load_async_to_lds_b128((gi32x4*)gsrc, (li32x4*)ldst, 0, 0);
      __builtin_amdgcn_global_load_async_to_lds_b128((gi32x4*)(gsrc + 8), (li32x4*)(ldst + 8), 0, 0);
    } else
#endif
    if (A_BF16) {
      const u32* src = (const u32*)((const u16*)Av + (size_t)(m0 + arow) * Cdim + kk + acol);
      u32* dst = (u32*)&As[arow * 32 + acol];
      #pragma unroll
      for (int i = 0; i < 8; ++i) dst[i] = src[i];
    } else {
      const float* src = (const float*)Av + (size_t)(m0 + arow) * Cdim + kk + acol;
      u32 packed[8];
      #pragma unroll
      for (int i = 0; i < 8; ++i)
        packed[i] = pack2bf(src[2 * i], src[2 * i + 1]);
      u32* dst = (u32*)&As[arow * 32 + acol];
      #pragma unroll
      for (int i = 0; i < 8; ++i) dst[i] = packed[i];
    }
    // ---- stage W tile (32 x 128) transposed -> Bs[n][k]
    {
      const float* src = W + (size_t)(kk + brow) * Cdim + n0 + bcol;
      #pragma unroll
      for (int i = 0; i < 8; ++i) {
        const u32 p = pack2bf(src[2 * i], src[2 * i + 1]);
        Bs[(bcol + 2 * i)     * 32 + brow] = (u16)p;
        Bs[(bcol + 2 * i + 1) * 32 + brow] = (u16)(p >> 16);
      }
    }
#if USE_ASYNC_LDS
    if (A_BF16) __builtin_amdgcn_s_wait_asynccnt(0);
#endif
    __syncthreads();

    // ---- A fragments (16x32 bf16): lane<16 holds K {0..7,16..23}, else {8..15,24..31}
    FragBF a[2];
    const int kbA = hi ? 8 : 0;
    #pragma unroll
    for (int im = 0; im < 2; ++im) {
      const int row = wm + im * 16 + ml;
      #pragma unroll
      for (int j = 0; j < 8; ++j) {
        const int k = kbA + ((j < 4) ? 2 * j : 2 * j + 8);
        a[im].u[j] = *(const u32*)&As[row * 32 + k];
      }
    }
    // ---- B fragments (32x16): lane<16 -> K {0..7,16..23}, lane>=16 -> K {8..15,24..31}, N = lane&15
    FragBF bf[4];
    const int kbB = hi ? 8 : 0;
    #pragma unroll
    for (int jn = 0; jn < 4; ++jn) {
      const u32* src = (const u32*)&Bs[(wn + jn * 16 + ml) * 32 + kbB];
      #pragma unroll
      for (int j = 0; j < 8; ++j) bf[jn].u[j] = src[(j < 4) ? j : j + 4];
    }
    #pragma unroll
    for (int im = 0; im < 2; ++im)
      #pragma unroll
      for (int jn = 0; jn < 4; ++jn)
        acc[im][jn] = __builtin_amdgcn_wmma_f32_16x16x32_bf16(
            false, a[im].v, false, bf[jn].v, (short)0, acc[im][jn], false, false);
  }

  // ---- epilogue: C/D layout VGPR r -> row r (lane<16) / r+8, col = lane&15
  #pragma unroll
  for (int im = 0; im < 2; ++im) {
    #pragma unroll
    for (int jn = 0; jn < 4; ++jn) {
      const int n = n0 + wn + jn * 16 + ml;
      const float bval = bias[n];
      #pragma unroll
      for (int r = 0; r < 8; ++r) {
        const int gm = m0 + wm + im * 16 + r + hi * 8;
        float val = acc[im][jn][r] + bval;
        if (MODE == MODE_O) {
          ((float*)out)[(size_t)gm * Cdim + n] = val;
        } else {
          if (MODE == MODE_Q) val *= 0.125f;   // 1/sqrt(DQK) folded into q
          const int bb = gm >> 11;             // / T
          const int tt = gm & (Tlen - 1);      // % T
          const int hh = n >> 6;
          const int dd = n & 63;
          ((u16*)out)[(((size_t)bb * Hn + hh) * Tlen + tt) * Dh + dd] = f2bf(val);
        }
      }
    }
  }
}

// ---------------------------------------------------------------------------
// Flash attention: one block per (query-tile of 64, head, batch).
// 4 waves x 16 query rows. Online softmax, S and O via bf16 WMMA.
// K tile staged by the Tensor Data Mover (TENSORcnt) when available,
// overlapping the VALU/LDS work of the manual V-transpose staging.
// ---------------------------------------------------------------------------
__global__ __launch_bounds__(128)
void flash_attn_kernel(const u16* __restrict__ qs, const u16* __restrict__ ks,
                       const u16* __restrict__ vs, u16* __restrict__ attn)
{
  __shared__ __align__(16) u16 Ks[64 * 64];     // [key][d]
  __shared__ __align__(16) u16 Vt[64 * 64];     // [d][key] (transposed)
  __shared__ __align__(16) u16 Ps[4][16 * 64];  // per-wave P tile [row][key]

  const int tid  = threadIdx.x;
  const int lane = tid & 31;
  const int w    = tid >> 5;
  const int ml   = lane & 15;
  const int hi   = (lane >> 4) & 1;
  const int h    = blockIdx.y;
  const int b    = blockIdx.z;
  const size_t base = ((size_t)b * Hn + h) * Tlen * Dh;
  const int q0   = blockIdx.x * 64 + w * 16;

  // Q fragments held for the whole kernel (DQK=64 -> two K=32 chunks)
  FragBF aq[2];
  const int kbA = hi ? 8 : 0;
  #pragma unroll
  for (int c = 0; c < 2; ++c)
    #pragma unroll
    for (int j = 0; j < 8; ++j) {
      const int d = c * 32 + kbA + ((j < 4) ? 2 * j : 2 * j + 8);
      aq[c].u[j] = *(const u32*)&qs[base + (size_t)(q0 + ml) * Dh + d];
    }

  v8f acco[4] = {};
  float mrow[8], lrow[8];
  #pragma unroll
  for (int r = 0; r < 8; ++r) { mrow[r] = -1e30f; lrow[r] = 0.f; }

  const int srow = tid >> 1;         // 0..63
  const int sd0  = (tid & 1) * 32;   // 0,32
  const int kbB  = hi ? 8 : 0;

  for (int jt = 0; jt <= (int)blockIdx.x; ++jt) {
    const int j0 = jt * 64;
    __syncthreads();

    // ---- stage K [key][d] (raw 64x64 bf16 tile copy)
#if USE_TDM
    if (w == 0)   // one TDM descriptor per workgroup; EXEC is ignored by TDM
      tdm_load_2d_b16(ks + base + (size_t)j0 * Dh, lds_off(&Ks[0]),
                      Dh, Tlen, Dh, 64, Dh);
#elif USE_ASYNC_LDS
    {
      const u16* gsrc = ks + base + (size_t)(j0 + srow) * Dh + sd0;
      u16* ldst = &Ks[srow * 64 + sd0];
      #pragma unroll
      for (int i = 0; i < 4; ++i)
        __builtin_amdgcn_global_load_async_to_lds_b128(
            (gi32x4*)(gsrc + i * 8), (li32x4*)(ldst + i * 8), 0, 0);
    }
#else
    {
      const uint4* ksrc = (const uint4*)(ks + base + (size_t)(j0 + srow) * Dh + sd0);
      uint4* kdst = (uint4*)&Ks[srow * 64 + sd0];
      #pragma unroll
      for (int i = 0; i < 4; ++i) kdst[i] = ksrc[i];
    }
#endif
    // ---- stage V transposed [d][key] (overlaps the TDM transfer)
    {
      union { uint4 q4[4]; u16 s[32]; } vtmp;
      const uint4* vsrc = (const uint4*)(vs + base + (size_t)(j0 + srow) * Dh + sd0);
      #pragma unroll
      for (int i = 0; i < 4; ++i) vtmp.q4[i] = vsrc[i];
      #pragma unroll
      for (int i = 0; i < 32; ++i) Vt[(sd0 + i) * 64 + srow] = vtmp.s[i];
    }
    if (jt < (int)blockIdx.x) {   // global_prefetch_b8 of next tile
      __builtin_prefetch(ks + base + (size_t)(j0 + 64 + srow) * Dh + sd0, 0, 1);
      __builtin_prefetch(vs + base + (size_t)(j0 + 64 + srow) * Dh + sd0, 0, 1);
    }
#if USE_TDM
    if (w == 0) __builtin_amdgcn_s_wait_tensorcnt(0);
#elif USE_ASYNC_LDS
    __builtin_amdgcn_s_wait_asynccnt(0);
#endif
    __syncthreads();

    // ---- S = q * K^T  (16 x 64 per wave), pre-scaled at q store time
    v8f sfrag[4];
    #pragma unroll
    for (int jn = 0; jn < 4; ++jn) {
      v8f c = {};
      #pragma unroll
      for (int kc = 0; kc < 2; ++kc) {
        FragBF bk;
        const u32* src = (const u32*)&Ks[(jn * 16 + ml) * 64 + kc * 32 + kbB];
        #pragma unroll
        for (int j = 0; j < 8; ++j) bk.u[j] = src[(j < 4) ? j : j + 4];
        c = __builtin_amdgcn_wmma_f32_16x16x32_bf16(false, aq[kc].v, false, bk.v,
                                                    (short)0, c, false, false);
      }
      sfrag[jn] = c;
    }

    // ---- causal mask on the diagonal tile
    if (jt == (int)blockIdx.x) {
      #pragma unroll
      for (int jn = 0; jn < 4; ++jn) {
        const int key = j0 + jn * 16 + ml;
        #pragma unroll
        for (int r = 0; r < 8; ++r)
          if (key > q0 + r + hi * 8) sfrag[jn][r] = -1e30f;
      }
    }

    // ---- online softmax (row stats via 16-lane xor-shuffle reductions)
    #pragma unroll
    for (int r = 0; r < 8; ++r) {
      float mx = fmaxf(fmaxf(sfrag[0][r], sfrag[1][r]),
                       fmaxf(sfrag[2][r], sfrag[3][r]));
      #pragma unroll
      for (int s = 1; s < 16; s <<= 1) mx = fmaxf(mx, __shfl_xor(mx, s, 32));
      const float mnew  = fmaxf(mrow[r], mx);
      const float alpha = __expf(mrow[r] - mnew);
      float rs = 0.f;
      #pragma unroll
      for (int jn = 0; jn < 4; ++jn) {
        const float p = __expf(sfrag[jn][r] - mnew);
        sfrag[jn][r] = p;
        rs += p;
      }
      #pragma unroll
      for (int s = 1; s < 16; s <<= 1) rs += __shfl_xor(rs, s, 32);
      lrow[r] = lrow[r] * alpha + rs;
      mrow[r] = mnew;
      #pragma unroll
      for (int jn = 0; jn < 4; ++jn) acco[jn][r] *= alpha;
    }

    // ---- P -> LDS (C-layout -> A-layout conversion through shared memory)
    #pragma unroll
    for (int jn = 0; jn < 4; ++jn)
      #pragma unroll
      for (int r = 0; r < 8; ++r)
        Ps[w][(r + hi * 8) * 64 + jn * 16 + ml] = f2bf(sfrag[jn][r]);
    __syncthreads();

    // ---- O += P * V
    #pragma unroll
    for (int kc = 0; kc < 2; ++kc) {
      FragBF ap;
      #pragma unroll
      for (int j = 0; j < 8; ++j) {
        const int k = kc * 32 + kbA + ((j < 4) ? 2 * j : 2 * j + 8);
        ap.u[j] = *(const u32*)&Ps[w][ml * 64 + k];
      }
      #pragma unroll
      for (int jn = 0; jn < 4; ++jn) {
        FragBF bv;
        const u32* src = (const u32*)&Vt[(jn * 16 + ml) * 64 + kc * 32 + kbB];
        #pragma unroll
        for (int j = 0; j < 8; ++j) bv.u[j] = src[(j < 4) ? j : j + 4];
        acco[jn] = __builtin_amdgcn_wmma_f32_16x16x32_bf16(false, ap.v, false, bv.v,
                                                           (short)0, acco[jn], false, false);
      }
    }
  }

  // ---- normalize and store bf16 to attn buffer [b][t][h*64+d]
  #pragma unroll
  for (int jn = 0; jn < 4; ++jn) {
    #pragma unroll
    for (int r = 0; r < 8; ++r) {
      const int tt = q0 + r + hi * 8;
      const int dd = jn * 16 + ml;
      const float val = acco[jn][r] / lrow[r];
      attn[((size_t)b * Tlen + tt) * (Hn * Dh) + h * Dh + dd] = f2bf(val);
    }
  }
}

// ---------------------------------------------------------------------------
extern "C" void kernel_launch(void* const* d_in, const int* in_sizes, int n_in,
                              void* d_out, int out_size, void* d_ws, size_t ws_size,
                              hipStream_t stream)
{
  (void)in_sizes; (void)n_in; (void)out_size; (void)ws_size;
  const float* Q  = (const float*)d_in[0];
  const float* K  = (const float*)d_in[1];
  const float* V  = (const float*)d_in[2];
  // d_in[3] = causal mask (bool) — structure is known, not read
  const float* Wq = (const float*)d_in[4];
  const float* bq = (const float*)d_in[5];
  const float* Wk = (const float*)d_in[6];
  const float* bk = (const float*)d_in[7];
  const float* Wv = (const float*)d_in[8];
  const float* bv = (const float*)d_in[9];
  const float* Wo = (const float*)d_in[10];
  const float* bo = (const float*)d_in[11];

  const size_t elems = (size_t)Bsz * Hn * Tlen * Dh;  // 4 Mi bf16 each
  u16* qs   = (u16*)d_ws;
  u16* ks   = qs + elems;
  u16* vs   = ks + elems;
  u16* attn = vs + elems;                              // 32 MB total workspace

  dim3 gg(Cdim / 128, (Bsz * Tlen) / 128);             // (8, 32)
  gemm_bias_kernel<0, MODE_Q><<<gg, 256, 0, stream>>>(Q, Wq, bq, qs);
  gemm_bias_kernel<0, MODE_K><<<gg, 256, 0, stream>>>(K, Wk, bk, ks);
  gemm_bias_kernel<0, MODE_V><<<gg, 256, 0, stream>>>(V, Wv, bv, vs);
  flash_attn_kernel<<<dim3(Tlen / 64, Hn, Bsz), 128, 0, stream>>>(qs, ks, vs, attn);
  gemm_bias_kernel<1, MODE_O><<<gg, 256, 0, stream>>>(attn, Wo, bo, d_out);
}
